// Attention_52536039965180
// MI455X (gfx1250) — matrix operand order repaired and checked by reading, not compiled, not run on hardware
//
#include <hip/hip_runtime.h>
#include <hip/hip_bf16.h>

typedef __attribute__((ext_vector_type(16))) __bf16 v16bf;
typedef __attribute__((ext_vector_type(8)))  __bf16 v8bf;
typedef __attribute__((ext_vector_type(8)))  float  v8f;

union Frag { v16bf v; v8bf h[2]; };

static __device__ __forceinline__ __bf16 f2bf(float f) {
    unsigned u = __float_as_uint(f);
    unsigned r = (u + 0x7FFFu + ((u >> 16) & 1u)) >> 16;
    unsigned short s = (unsigned short)r;
    __bf16 b;
    __builtin_memcpy(&b, &s, 2);
    return b;
}

// ---------------- elementwise f32 -> bf16 ----------------
__global__ void cvt_kernel(const float* __restrict__ X, __bf16* __restrict__ Xh, int n) {
    int i = blockIdx.x * blockDim.x + threadIdx.x;
    if (i < n) Xh[i] = f2bf(X[i]);
}

// --------- transpose + convert: Wt[n*K+k] = bf16(W[k*NT+n]) ---------
__global__ void tcvt_kernel(const float* __restrict__ W, __bf16* __restrict__ Wt,
                            int K, int NT, int total) {
    int idx = blockIdx.x * blockDim.x + threadIdx.x;
    if (idx >= total) return;
    int n = idx / K;
    int k = idx - n * K;
    Wt[idx] = f2bf(W[(size_t)k * NT + n]);
}

// ---------------- bf16 WMMA GEMM: C = A @ Bt^T + bias ----------------
// A: [M x K] row-major bf16; Bt: [N x K] row-major bf16 (i.e. B transposed)
// Block = 256 threads (8 waves), block tile 128x128, wave tile 32x64.
__global__ __launch_bounds__(256) void gemm_bf16_kernel(
    const __bf16* __restrict__ A, const __bf16* __restrict__ Bt,
    const float* __restrict__ bias, float* __restrict__ Cmat,
    int M, int Nn, int K) {
    const int lane = threadIdx.x & 31;
    const int wave = threadIdx.x >> 5;
    const int wr = wave >> 1, wc = wave & 1;
    const int m16 = lane & 15, hi = lane >> 4;
    const int row0 = blockIdx.y * 128 + wr * 32;
    const int col0 = blockIdx.x * 128 + wc * 64;

    v8f acc[2][4];
#pragma unroll
    for (int tr = 0; tr < 2; ++tr)
#pragma unroll
        for (int tc = 0; tc < 4; ++tc)
#pragma unroll
            for (int r = 0; r < 8; ++r) acc[tr][tc][r] = 0.0f;

    for (int k0 = 0; k0 < K; k0 += 32) {
        Frag a[2], b[4];
#pragma unroll
        for (int tr = 0; tr < 2; ++tr) {
            // A fragment (16-bit A 16x32 layout): lane m=lane%16 holds row,
            // elems 0..7 -> K = k0 + hi*8 + e, elems 8..15 -> +16
            const __bf16* p = A + (size_t)(row0 + tr * 16 + m16) * K + k0 + hi * 8;
            a[tr].h[0] = *(const v8bf*)p;
            a[tr].h[1] = *(const v8bf*)(p + 16);
        }
#pragma unroll
        for (int tc = 0; tc < 4; ++tc) {
            // B fragment (32x16): lane n=lane%16 is a column (a row of Bt),
            // elems 0..7 -> K = k0 + hi*8 + e, elems 8..15 -> +16
            const __bf16* p = Bt + (size_t)(col0 + tc * 16 + m16) * K + k0 + hi * 8;
            b[tc].h[0] = *(const v8bf*)p;
            b[tc].h[1] = *(const v8bf*)(p + 16);
        }
#pragma unroll
        for (int tr = 0; tr < 2; ++tr)
#pragma unroll
            for (int tc = 0; tc < 4; ++tc)
                acc[tr][tc] = __builtin_amdgcn_wmma_f32_16x16x32_bf16(
                    false, a[tr].v, false, b[tc].v, (short)0, acc[tr][tc], false, false);
    }

#pragma unroll
    for (int tr = 0; tr < 2; ++tr)
#pragma unroll
        for (int tc = 0; tc < 4; ++tc)
#pragma unroll
            for (int r = 0; r < 8; ++r) {
                int row = row0 + tr * 16 + r + hi * 8;
                int col = col0 + tc * 16 + m16;
                Cmat[(size_t)row * Nn + col] = acc[tr][tc][r] + bias[col];
            }
}

// ---------------- fused flash attention with RMSNorm ----------------
// grid = (N/128, H, B); block = 256 (8 waves); each wave owns 16 q rows.
#define ATT_N 2048
#define ATT_C 1024
#define ATT_3C 3072
#define ATT_HD 64
__global__ __launch_bounds__(256) void attn_kernel(
    const float* __restrict__ qkv, const float* __restrict__ qnw,
    const float* __restrict__ knw, __bf16* __restrict__ Ah) {
    const int qb = blockIdx.x, h = blockIdx.y, b = blockIdx.z;
    const int tid = threadIdx.x;
    const int lane = tid & 31, wave = tid >> 5;
    const int m16 = lane & 15, hi = lane >> 4;

    __shared__ __align__(16) __bf16 Qs[128 * 64];   // 16 KB  [qrow][d]
    __shared__ __align__(16) __bf16 Ks[64 * 64];    //  8 KB  [key][d]
    __shared__ __align__(16) __bf16 Vts[64 * 64];   //  8 KB  [d][key]
    __shared__ __align__(16) __bf16 Ps[8 * 16 * 64];// 16 KB  per-wave P tiles

    const size_t rowbase = (size_t)b * ATT_N;
    const int qcol = h * ATT_HD;
    const int kcol = ATT_C + h * ATT_HD;
    const int vcol = 2 * ATT_C + h * ATT_HD;

    // ---- load + RMSNorm Q block (threads 0..127, one row each) ----
    if (tid < 128) {
        int qrow = qb * 128 + tid;
        const float* p = qkv + (rowbase + qrow) * ATT_3C + qcol;
        float v[64];
        float ss = 0.f;
#pragma unroll
        for (int i = 0; i < 16; ++i) {
            float4 f = ((const float4*)p)[i];
            v[4 * i + 0] = f.x; v[4 * i + 1] = f.y;
            v[4 * i + 2] = f.z; v[4 * i + 3] = f.w;
            ss += f.x * f.x + f.y * f.y + f.z * f.z + f.w * f.w;
        }
        float nrm = rsqrtf(ss * (1.0f / 64.0f) + 1e-6f);
#pragma unroll
        for (int d = 0; d < 64; ++d) Qs[tid * 64 + d] = f2bf(v[d] * nrm * qnw[d]);
    }

    v8f Oacc[4];
    float mrow[8], lrow[8];
#pragma unroll
    for (int t = 0; t < 4; ++t)
#pragma unroll
        for (int r = 0; r < 8; ++r) Oacc[t][r] = 0.0f;
#pragma unroll
    for (int r = 0; r < 8; ++r) { mrow[r] = -1e30f; lrow[r] = 0.0f; }

    for (int kt = 0; kt < ATT_N / 64; ++kt) {
        __syncthreads();
        if (tid < 64) {
            // K tile row with RMSNorm
            int krow = kt * 64 + tid;
            const float* p = qkv + (rowbase + krow) * ATT_3C + kcol;
            float v[64];
            float ss = 0.f;
#pragma unroll
            for (int i = 0; i < 16; ++i) {
                float4 f = ((const float4*)p)[i];
                v[4 * i + 0] = f.x; v[4 * i + 1] = f.y;
                v[4 * i + 2] = f.z; v[4 * i + 3] = f.w;
                ss += f.x * f.x + f.y * f.y + f.z * f.z + f.w * f.w;
            }
            float nrm = rsqrtf(ss * (1.0f / 64.0f) + 1e-6f);
#pragma unroll
            for (int d = 0; d < 64; ++d) Ks[tid * 64 + d] = f2bf(v[d] * nrm * knw[d]);
        } else if (tid < 128) {
            // V tile, stored transposed: Vts[d][key]
            int key = tid - 64;
            int krow = kt * 64 + key;
            const float* p = qkv + (rowbase + krow) * ATT_3C + vcol;
#pragma unroll
            for (int i = 0; i < 16; ++i) {
                float4 f = ((const float4*)p)[i];
                Vts[(4 * i + 0) * 64 + key] = f2bf(f.x);
                Vts[(4 * i + 1) * 64 + key] = f2bf(f.y);
                Vts[(4 * i + 2) * 64 + key] = f2bf(f.z);
                Vts[(4 * i + 3) * 64 + key] = f2bf(f.w);
            }
        }
        __syncthreads();

        // ---- S = (Q Kt^T) * scale, wave owns 16 q rows x 64 keys ----
        v8f S[4];
#pragma unroll
        for (int t = 0; t < 4; ++t)
#pragma unroll
            for (int r = 0; r < 8; ++r) S[t][r] = 0.0f;
#pragma unroll
        for (int ks = 0; ks < 2; ++ks) {
            Frag a;
            const __bf16* ab = &Qs[(wave * 16 + m16) * 64 + ks * 32 + hi * 8];
            a.h[0] = *(const v8bf*)ab;
            a.h[1] = *(const v8bf*)(ab + 16);
#pragma unroll
            for (int t = 0; t < 4; ++t) {
                Frag bb;
                const __bf16* bp = &Ks[(t * 16 + m16) * 64 + ks * 32 + hi * 8];
                bb.h[0] = *(const v8bf*)bp;
                bb.h[1] = *(const v8bf*)(bp + 16);
                S[t] = __builtin_amdgcn_wmma_f32_16x16x32_bf16(
                    false, a.v, false, bb.v, (short)0, S[t], false, false);
            }
        }

        // ---- online softmax; row lives across 16 lanes of a half-wave ----
#pragma unroll
        for (int r = 0; r < 8; ++r) {
            float mx = -1e30f;
#pragma unroll
            for (int t = 0; t < 4; ++t) {
                S[t][r] *= 0.125f;            // 1/sqrt(64)
                mx = fmaxf(mx, S[t][r]);
            }
#pragma unroll
            for (int off = 1; off < 16; off <<= 1)
                mx = fmaxf(mx, __shfl_xor(mx, off, 16));
            float mnew = fmaxf(mrow[r], mx);
            float alpha = __expf(mrow[r] - mnew);
            float sum = 0.f;
#pragma unroll
            for (int t = 0; t < 4; ++t) {
                float pv = __expf(S[t][r] - mnew);
                S[t][r] = pv;
                sum += pv;
            }
#pragma unroll
            for (int off = 1; off < 16; off <<= 1)
                sum += __shfl_xor(sum, off, 16);
            lrow[r] = lrow[r] * alpha + sum;
            mrow[r] = mnew;
#pragma unroll
            for (int t = 0; t < 4; ++t) Oacc[t][r] *= alpha;
        }

        // ---- stage P (bf16) in per-wave LDS for A-fragment re-layout ----
        __bf16* pw = &Ps[wave * 16 * 64];
#pragma unroll
        for (int t = 0; t < 4; ++t)
#pragma unroll
            for (int r = 0; r < 8; ++r)
                pw[(r + hi * 8) * 64 + t * 16 + m16] = f2bf(S[t][r]);

        // ---- O += P @ V ----
#pragma unroll
        for (int ks = 0; ks < 2; ++ks) {
            Frag a;
            const __bf16* ab = pw + m16 * 64 + ks * 32 + hi * 8;
            a.h[0] = *(const v8bf*)ab;
            a.h[1] = *(const v8bf*)(ab + 16);
#pragma unroll
            for (int t = 0; t < 4; ++t) {
                Frag bb;
                const __bf16* bp = &Vts[(t * 16 + m16) * 64 + ks * 32 + hi * 8];
                bb.h[0] = *(const v8bf*)bp;
                bb.h[1] = *(const v8bf*)(bp + 16);
                Oacc[t] = __builtin_amdgcn_wmma_f32_16x16x32_bf16(
                    false, a.v, false, bb.v, (short)0, Oacc[t], false, false);
            }
        }
    }

    // ---- epilogue: O /= l, write bf16 attn output [b,n,h,d] -> [(b*N+n), C] ----
#pragma unroll
    for (int r = 0; r < 8; ++r) {
        float inv = 1.0f / lrow[r];
        int qrow = qb * 128 + wave * 16 + r + hi * 8;
        size_t orow = (rowbase + qrow) * ATT_C + h * ATT_HD;
#pragma unroll
        for (int t = 0; t < 4; ++t)
            Ah[orow + t * 16 + m16] = f2bf(Oacc[t][r] * inv);
    }
}

extern "C" void kernel_launch(void* const* d_in, const int* in_sizes, int n_in,
                              void* d_out, int out_size, void* d_ws, size_t ws_size,
                              hipStream_t stream) {
    const float* x      = (const float*)d_in[0];
    const float* qkv_w  = (const float*)d_in[1];
    const float* qkv_b  = (const float*)d_in[2];
    const float* qnw    = (const float*)d_in[3];
    const float* knw    = (const float*)d_in[4];
    const float* proj_w = (const float*)d_in[5];
    const float* proj_b = (const float*)d_in[6];
    float* out = (float*)d_out;

    char* ws = (char*)d_ws;
    __bf16* Xh     = (__bf16*)(ws);                        //  8 MB: x bf16 [4096x1024]
    __bf16* Wqkvt  = (__bf16*)(ws + (size_t)8  * 1024 * 1024);  //  6 MB: qkv_w^T bf16 [3072x1024]
    __bf16* Wprojt = (__bf16*)(ws + (size_t)14 * 1024 * 1024);  //  2 MB: proj_w^T bf16 [1024x1024]
    float*  QKV    = (float*) (ws + (size_t)16 * 1024 * 1024);  // 48 MB: qkv f32 [4096x3072]
    __bf16* Ah     = (__bf16*)(ws + (size_t)64 * 1024 * 1024);  //  8 MB: attn out bf16 [4096x1024]

    const int M = 4096, C = 1024, C3 = 3072;

    cvt_kernel<<<(M * C + 255) / 256, 256, 0, stream>>>(x, Xh, M * C);
    tcvt_kernel<<<(C3 * C + 255) / 256, 256, 0, stream>>>(qkv_w, Wqkvt, C, C3, C3 * C);
    tcvt_kernel<<<(C * C + 255) / 256, 256, 0, stream>>>(proj_w, Wprojt, C, C, C * C);

    // QKV = Xh @ qkv_w + qkv_b   (4096 x 3072)
    gemm_bf16_kernel<<<dim3(C3 / 128, M / 128), 256, 0, stream>>>(
        Xh, Wqkvt, qkv_b, QKV, M, C3, C);

    // fused RMSNorm + flash attention  (grid: qblocks x heads x batch)
    attn_kernel<<<dim3(2048 / 128, 16, 2), 256, 0, stream>>>(QKV, qnw, knw, Ah);

    // out = Ah @ proj_w + proj_b  (4096 x 1024)
    gemm_bf16_kernel<<<dim3(C / 128, M / 128), 256, 0, stream>>>(
        Ah, Wprojt, proj_b, out, M, C, C);
}
